// MixtureOfExperts_88665304859118
// MI455X (gfx1250) — matrix operand order repaired and checked by reading, not compiled, not run on hardware
//
#include <hip/hip_runtime.h>
#include <hip/hip_bf16.h>
#include <math.h>

// Problem constants (from reference)
#define B 8192
#define D 1024
#define H 4096
#define O 1024
#define E 8

// GEMM tiling: 128x128 tile per 256-thread block (8 waves, 2x4),
// each wave owns a 64x32 sub-tile = 4x2 WMMA accumulators.
#define BM 128
#define BN 128
#define KS 32    // WMMA K step (bf16 16x16x32)
#define KSP 40   // padded LDS row stride (bank spread, keeps 16B alignment)

typedef __attribute__((ext_vector_type(16))) __bf16 v16bf;
typedef __attribute__((ext_vector_type(8)))  float  v8f;

// Pointer payload/address-space types for the async-to-LDS builtin
// (diagnostic-confirmed: param 0 is v4i in AS1 / "__device__").
typedef int v4i __attribute__((__vector_size__(16)));
typedef __attribute__((address_space(1))) v4i gv4i_t;
typedef __attribute__((address_space(3))) v4i lv4i_t;

#if defined(__has_builtin)
#if __has_builtin(__builtin_amdgcn_global_load_async_to_lds_b128) && \
    __has_builtin(__builtin_amdgcn_s_wait_asynccnt)
#define USE_ASYNC 1
#endif
#endif
#ifndef USE_ASYNC
#define USE_ASYNC 0
#endif

__device__ __forceinline__ v8f wmma_bf16(v16bf a, v16bf b, v8f c) {
  return __builtin_amdgcn_wmma_f32_16x16x32_bf16(
      /*neg_a=*/false, a, /*neg_b=*/false, b,
      /*c_mod=*/(short)0, c, /*reuse_a=*/false, /*reuse_b=*/false);
}

// 16-byte global -> LDS copy; async (no VGPR round trip) when available.
__device__ __forceinline__ void copy16_g2l(const __bf16* gsrc, __bf16* ldst) {
#if USE_ASYNC
  __builtin_amdgcn_global_load_async_to_lds_b128((gv4i_t*)gsrc,
                                                 (lv4i_t*)ldst, 0, 0);
#else
  *(uint4*)ldst = *(const uint4*)gsrc;
#endif
}

__device__ __forceinline__ void wait_async_group_inflight() {
#if USE_ASYNC
  __builtin_amdgcn_s_wait_asynccnt(4);  // next slice (4 per wave) may fly
#endif
}
__device__ __forceinline__ void wait_async_all() {
#if USE_ASYNC
  __builtin_amdgcn_s_wait_asynccnt(0);
#endif
}

// ---------------------------------------------------------------------------
// zero helper (harness poisons d_out / d_ws)
// ---------------------------------------------------------------------------
__global__ void zero_kernel(float* __restrict__ p, size_t n) {
  size_t i = (size_t)blockIdx.x * blockDim.x + threadIdx.x;
  size_t stride = (size_t)gridDim.x * blockDim.x;
  for (size_t j = i; j < n; j += stride) p[j] = 0.0f;
}

// ---------------------------------------------------------------------------
// fp32 -> bf16 conversion (row-major copy, for activations)
// ---------------------------------------------------------------------------
__global__ void f32_to_bf16_kernel(const float* __restrict__ s,
                                   __bf16* __restrict__ d, size_t n) {
  size_t i = ((size_t)blockIdx.x * blockDim.x + threadIdx.x) * 4;
  const size_t stride = (size_t)gridDim.x * blockDim.x * 4;
  for (; i + 3 < n; i += stride) {
    const float4 v = *(const float4*)(s + i);
    d[i + 0] = (__bf16)v.x;
    d[i + 1] = (__bf16)v.y;
    d[i + 2] = (__bf16)v.z;
    d[i + 3] = (__bf16)v.w;
  }
}

// ---------------------------------------------------------------------------
// fp32 -> bf16 conversion with transpose (for weights, so GEMM B operands
// are K-contiguous and B fragments load as 2x ds_load_b128 like A).
// src: R x C row-major fp32; dst: C x R row-major bf16. 32x32 LDS tiles.
// ---------------------------------------------------------------------------
__global__ __launch_bounds__(256) void convert_transpose_kernel(
    const float* __restrict__ src, __bf16* __restrict__ dst, int R, int C) {
  __shared__ float tile[32][33];
  const int c0 = blockIdx.x * 32;
  const int r0 = blockIdx.y * 32;
  const int tx = threadIdx.x;  // 0..31
  const int ty = threadIdx.y;  // 0..7
  for (int i = ty; i < 32; i += 8)
    tile[i][tx] = src[(size_t)(r0 + i) * C + c0 + tx];
  __syncthreads();
  for (int i = ty; i < 32; i += 8)
    dst[(size_t)(c0 + i) * R + r0 + tx] = (__bf16)tile[tx][i];
}

// ---------------------------------------------------------------------------
// Gate: logits = x @ gate_W + gate_b, softmax, top-2, renormalized weights.
// ---------------------------------------------------------------------------
__global__ __launch_bounds__(256) void gate_kernel(
    const float* __restrict__ x, const float* __restrict__ gW,
    const float* __restrict__ gb, float* __restrict__ sel_w,
    float* __restrict__ counts, float* __restrict__ usage) {
  const int wave = threadIdx.x >> 5;
  const int lane = threadIdx.x & 31;
  const int t = blockIdx.x * 8 + wave;
  if (t >= B) return;

  float acc[E];
#pragma unroll
  for (int e = 0; e < E; ++e) acc[e] = 0.0f;

  const float* xr = x + (size_t)t * D;
  for (int d = lane; d < D; d += 32) {
    const float xv = xr[d];
    const float* g = gW + (size_t)d * E;
#pragma unroll
    for (int e = 0; e < E; ++e) acc[e] += xv * g[e];
  }
#pragma unroll
  for (int e = 0; e < E; ++e) {
#pragma unroll
    for (int off = 16; off > 0; off >>= 1)
      acc[e] += __shfl_xor(acc[e], off, 32);
  }

  if (lane == 0) {
    float l[E];
    float m = -1e30f;
#pragma unroll
    for (int e = 0; e < E; ++e) { l[e] = acc[e] + gb[e]; m = fmaxf(m, l[e]); }
    float s = 0.0f;
#pragma unroll
    for (int e = 0; e < E; ++e) { l[e] = __expf(l[e] - m); s += l[e]; }
    const float inv = 1.0f / s;
    float g[E];
#pragma unroll
    for (int e = 0; e < E; ++e) g[e] = l[e] * inv;

    int i0 = 0;
#pragma unroll
    for (int e = 1; e < E; ++e) if (g[e] > g[i0]) i0 = e;
    int i1 = (i0 == 0) ? 1 : 0;
#pragma unroll
    for (int e = 0; e < E; ++e) if (e != i0 && g[e] > g[i1]) i1 = e;

    const float denom = g[i0] + g[i1];
    const float w0 = g[i0] / denom, w1 = g[i1] / denom;
#pragma unroll
    for (int e = 0; e < E; ++e)
      sel_w[(size_t)t * E + e] = (e == i0) ? w0 : ((e == i1) ? w1 : 0.0f);
    atomicAdd(&counts[i0], 1.0f);
    atomicAdd(&counts[i1], 1.0f);
#pragma unroll
    for (int e = 0; e < E; ++e) atomicAdd(&usage[e], g[e]);
  }
}

// ---------------------------------------------------------------------------
// Stage one (BM x KS) A slice and (BN x KS) B slice into LDS.
// Both operands are K-contiguous (A row-major MxK, B transposed NxK).
// Each wave issues exactly 4 copy instructions (2 A + 2 B).
// ---------------------------------------------------------------------------
__device__ __forceinline__ void stage_tiles(const __bf16* __restrict__ Ag,
                                            int lda,
                                            const __bf16* __restrict__ BgT,
                                            int ldb, int k0,
                                            __bf16 (*la)[KSP],
                                            __bf16 (*lb)[KSP], int tid) {
#pragma unroll
  for (int i = tid; i < (BM * KS) / 8; i += 256) {
    const int r = i >> 2, c8 = (i & 3) * 8;
    copy16_g2l(Ag + (size_t)r * lda + k0 + c8, &la[r][c8]);
  }
#pragma unroll
  for (int i = tid; i < (BN * KS) / 8; i += 256) {
    const int r = i >> 2, c8 = (i & 3) * 8;
    copy16_g2l(BgT + (size_t)r * ldb + k0 + c8, &lb[r][c8]);
  }
}

// ---------------------------------------------------------------------------
// WMMA GEMM core with double-buffered LDS + async global->LDS staging:
// acc += A(BM x Kdim, row-major, ld=lda) @ B(Kdim x BN given as BN x Kdim
// transposed, ld=ldb). Ag/BgT pre-offset to the tile base.
// ---------------------------------------------------------------------------
__device__ __forceinline__ void gemm_core(const __bf16* __restrict__ Ag,
                                          int lda,
                                          const __bf16* __restrict__ BgT,
                                          int ldb, int Kdim,
                                          __bf16 (*lds_a)[BM][KSP],
                                          __bf16 (*lds_b)[BN][KSP],
                                          v8f acc[4][2], int tid, int wave,
                                          int lane) {
  const int wr = wave >> 2;  // 0..1 : 64-row group
  const int wc = wave & 3;   // 0..3 : 32-col group
  const int Ml = lane & 15;
  const int hi = lane >> 4;
  const int off8 = hi * 8;
  const int koff16 = hi * 16;
  const int Nl = lane & 15;

  stage_tiles(Ag, lda, BgT, ldb, 0, lds_a[0], lds_b[0], tid);

  int cur = 0;
#pragma unroll 1
  for (int k0 = 0; k0 < Kdim; k0 += KS) {
    const bool has_next = (k0 + KS) < Kdim;
    if (has_next)
      stage_tiles(Ag, lda, BgT, ldb, k0 + KS, lds_a[cur ^ 1], lds_b[cur ^ 1],
                  tid);
    if (has_next)
      wait_async_group_inflight();  // current slice done; next may fly
    else
      wait_async_all();
    __syncthreads();

    v16bf a[4], bfr[2];
#pragma unroll
    for (int mt = 0; mt < 4; ++mt) {
      const __bf16* rp = &lds_a[cur][wr * 64 + mt * 16 + Ml][0];
#pragma unroll
      for (int j = 0; j < 8; ++j) {
        a[mt][j]     = rp[off8 + j];
        a[mt][8 + j] = rp[16 + off8 + j];
      }
    }
#pragma unroll
    for (int nt = 0; nt < 2; ++nt) {
      const __bf16* cp = &lds_b[cur][wc * 32 + nt * 16 + Nl][0];
#pragma unroll
      for (int j = 0; j < 16; ++j) bfr[nt][j] = cp[(j < 8 ? off8 + j : 8 + off8 + j)];
    }
#pragma unroll
    for (int mt = 0; mt < 4; ++mt)
#pragma unroll
      for (int nt = 0; nt < 2; ++nt)
        acc[mt][nt] = wmma_bf16(a[mt], bfr[nt], acc[mt][nt]);
    __syncthreads();  // everyone done reading buf[cur] before it is refilled
    cur ^= 1;
  }
}

// ---------------------------------------------------------------------------
// GEMM1: hbf = relu(xbf @ W1[e] + b1[e])  (B x H bf16 to workspace)
// Grid: (B/BM, H/BN). W1 given transposed (H x D bf16).
// ---------------------------------------------------------------------------
__global__ __launch_bounds__(256) void gemm1_kernel(
    const __bf16* __restrict__ xbf, const __bf16* __restrict__ w1T,
    const float* __restrict__ b1e, __bf16* __restrict__ hbf) {
  const int rowbase = blockIdx.x * BM;
  const int colbase = blockIdx.y * BN;

  __shared__ __bf16 lds_a[2][BM][KSP];
  __shared__ __bf16 lds_b[2][BN][KSP];

  const int tid = threadIdx.x, wave = tid >> 5, lane = tid & 31;
  const int wr = wave >> 2, wc = wave & 3;
  const int hi = lane >> 4, Nl = lane & 15;

  v8f acc[4][2];
  const v8f vz = {0.f, 0.f, 0.f, 0.f, 0.f, 0.f, 0.f, 0.f};
#pragma unroll
  for (int mt = 0; mt < 4; ++mt)
#pragma unroll
    for (int nt = 0; nt < 2; ++nt) acc[mt][nt] = vz;

  gemm_core(xbf + (size_t)rowbase * D, D, w1T + (size_t)colbase * D, D, D,
            lds_a, lds_b, acc, tid, wave, lane);

#pragma unroll
  for (int mt = 0; mt < 4; ++mt)
#pragma unroll
    for (int nt = 0; nt < 2; ++nt)
#pragma unroll
      for (int v = 0; v < 8; ++v) {
        const int rl = wr * 64 + mt * 16 + (v + 8 * hi);
        const int cl = wc * 32 + nt * 16 + Nl;
        const float hv = acc[mt][nt][v] + b1e[colbase + cl];
        hbf[(size_t)(rowbase + rl) * H + colbase + cl] =
            (__bf16)(hv > 0.0f ? hv : 0.0f);
      }
}

// ---------------------------------------------------------------------------
// GEMM2: out_e = hbf @ W2[e] + b2[e]; combine into final via sel_w and
// accumulate per-expert output sums. W2 given transposed (O x H bf16).
// Grid: (B/BM, O/BN).
// ---------------------------------------------------------------------------
__global__ __launch_bounds__(256) void gemm2_kernel(
    const __bf16* __restrict__ hbf, const __bf16* __restrict__ w2T,
    const float* __restrict__ b2e, const float* __restrict__ sel_w, int e,
    float* __restrict__ final_out, float* __restrict__ avg_e) {
  const int rowbase = blockIdx.x * BM;
  const int obase   = blockIdx.y * BN;

  __shared__ __bf16 lds_a[2][BM][KSP];
  __shared__ __bf16 lds_b[2][BN][KSP];

  const int tid = threadIdx.x, wave = tid >> 5, lane = tid & 31;
  const int wr = wave >> 2, wc = wave & 3;
  const int hi = lane >> 4, Nl = lane & 15;

  v8f acc[4][2];
  const v8f vz = {0.f, 0.f, 0.f, 0.f, 0.f, 0.f, 0.f, 0.f};
#pragma unroll
  for (int mt = 0; mt < 4; ++mt)
#pragma unroll
    for (int nt = 0; nt < 2; ++nt) acc[mt][nt] = vz;

  gemm_core(hbf + (size_t)rowbase * H, H, w2T + (size_t)obase * H, H, H,
            lds_a, lds_b, acc, tid, wave, lane);

#pragma unroll
  for (int mt = 0; mt < 4; ++mt)
#pragma unroll
    for (int nt = 0; nt < 2; ++nt)
#pragma unroll
      for (int v = 0; v < 8; ++v) {
        const int rl = wr * 64 + mt * 16 + (v + 8 * hi);
        const int cl = wc * 32 + nt * 16 + Nl;
        const int t = rowbase + rl;
        const int o = obase + cl;
        const float val = acc[mt][nt][v] + b2e[o];
        const float sw = sel_w[(size_t)t * E + e];
        if (sw > 0.0f) {  // token routed to this expert
          atomicAdd(&final_out[(size_t)t * O + o], sw * val);
          atomicAdd(&avg_e[o], val);
        }
      }
}

// ---------------------------------------------------------------------------
// Finalize: avgs = avg_sum / max(counts,1); distinct & equilibrium losses.
// ---------------------------------------------------------------------------
__global__ __launch_bounds__(256) void finalize_kernel(
    const float* __restrict__ counts, const float* __restrict__ usage,
    float* __restrict__ avg, float* __restrict__ out_tail) {
  __shared__ float red[256];
  const int tid = threadIdx.x;

  for (int i = tid; i < E * O; i += 256) {
    const int e = i / O;
    float c = counts[e];
    if (c < 1.0f) c = 1.0f;
    avg[i] = avg[i] / c;
  }
  __syncthreads();

  float total = 0.0f;  // meaningful on tid 0 only
  for (int a = 0; a < E; ++a) {
    for (int b = a + 1; b < E; ++b) {
      float part = 0.0f;
      for (int o = tid; o < O; o += 256) {
        const float d = avg[a * O + o] - avg[b * O + o];
        part += d * d;
      }
      red[tid] = part;
      __syncthreads();
      for (int s = 128; s > 0; s >>= 1) {
        if (tid < s) red[tid] += red[tid + s];
        __syncthreads();
      }
      if (tid == 0) {
        if (counts[a] > 0.0f && counts[b] > 0.0f)
          total += __expf(-red[0] * 0.5f);
      }
      __syncthreads();
    }
  }
  if (tid == 0) {
    out_tail[0] = total;
    float s = 0.0f;
#pragma unroll
    for (int e = 0; e < E; ++e) {
      const float u = usage[e] / (float)B - 1.0f / (float)E;
      s += u * u;
    }
    out_tail[1] = sqrtf(s);
  }
}

// ---------------------------------------------------------------------------
// Workspace layout (~96.5 MB):
//   sel_w B*E f32 | counts E | usage E | avg_sum E*O f32
//   xbf  B*D bf16 (16 MB)
//   hbf  B*H bf16 (64 MB, reused across experts)
//   w1T  H*D bf16 (8 MB, per-expert transposed weights)
//   w2T  O*H bf16 (8 MB, per-expert transposed weights)
// ---------------------------------------------------------------------------
extern "C" void kernel_launch(void* const* d_in, const int* in_sizes, int n_in,
                              void* d_out, int out_size, void* d_ws,
                              size_t ws_size, hipStream_t stream) {
  const float* x      = (const float*)d_in[0];
  const float* gate_W = (const float*)d_in[1];
  const float* gate_b = (const float*)d_in[2];
  const float* W1     = (const float*)d_in[3];
  const float* b1     = (const float*)d_in[4];
  const float* W2     = (const float*)d_in[5];
  const float* b2     = (const float*)d_in[6];
  // d_in[7] is k; K=2 is a compile-time constant of the reference.

  float* final_out = (float*)d_out;              // B*O
  float* loss_out  = final_out + (size_t)B * O;  // 2 scalars

  char* w = (char*)d_ws;
  float* sel_w   = (float*)w;  w += (size_t)B * E * sizeof(float);
  float* counts  = (float*)w;  w += E * sizeof(float);
  float* usage   = (float*)w;  w += E * sizeof(float);
  float* avg_sum = (float*)w;  w += (size_t)E * O * sizeof(float);
  __bf16* xbf = (__bf16*)w;    w += (size_t)B * D * sizeof(__bf16);
  __bf16* hbf = (__bf16*)w;    w += (size_t)B * H * sizeof(__bf16);
  __bf16* w1T = (__bf16*)w;    w += (size_t)D * H * sizeof(__bf16);
  __bf16* w2T = (__bf16*)w;    w += (size_t)H * O * sizeof(__bf16);

  zero_kernel<<<2048, 256, 0, stream>>>(final_out, (size_t)B * O);
  zero_kernel<<<32, 256, 0, stream>>>(counts, (size_t)(2 * E + E * O));

  gate_kernel<<<B / 8, 256, 0, stream>>>(x, gate_W, gate_b, sel_w, counts,
                                         usage);

  f32_to_bf16_kernel<<<2048, 256, 0, stream>>>(x, xbf, (size_t)B * D);

  const dim3 tdim(32, 8);
  const dim3 tgrid1(H / 32, D / 32);  // W1[e]: (D x H) -> (H x D)
  const dim3 tgrid2(O / 32, H / 32);  // W2[e]: (H x O) -> (O x H)
  const dim3 grid1(B / BM, H / BN);   // 64 x 32
  const dim3 grid2(B / BM, O / BN);   // 64 x 8
  for (int e = 0; e < E; ++e) {
    convert_transpose_kernel<<<tgrid1, tdim, 0, stream>>>(
        W1 + (size_t)e * D * H, w1T, D, H);
    convert_transpose_kernel<<<tgrid2, tdim, 0, stream>>>(
        W2 + (size_t)e * H * O, w2T, H, O);
    gemm1_kernel<<<grid1, 256, 0, stream>>>(xbf, w1T, b1 + (size_t)e * H, hbf);
    gemm2_kernel<<<grid2, 256, 0, stream>>>(hbf, w2T, b2 + (size_t)e * O,
                                            sel_w, e, final_out,
                                            avg_sum + (size_t)e * O);
  }

  finalize_kernel<<<1, 256, 0, stream>>>(counts, usage, avg_sum, loss_out);
}
